// GCNetwork_59193239273550
// MI455X (gfx1250) — compile-verified, matrix operand order checked by reading, not run on hardware
//
#include <hip/hip_runtime.h>

// ---------------------------------------------------------------------------
// Types for CDNA5 (gfx1250) WMMA / TDM
// ---------------------------------------------------------------------------
typedef __bf16 bf16;
typedef __attribute__((ext_vector_type(16))) __bf16        v16bf;
typedef __attribute__((ext_vector_type(8)))  float         v8f;
typedef __attribute__((ext_vector_type(4)))  unsigned int  u32x4;
typedef __attribute__((ext_vector_type(8)))  int           i32x8;
typedef __attribute__((ext_vector_type(4)))  int           i32x4;

#if defined(__has_builtin)
#  if __has_builtin(__builtin_amdgcn_tensor_load_to_lds)
#    define USE_TDM 1
#  endif
#endif
#ifndef USE_TDM
#  define USE_TDM 0
#endif

__device__ __forceinline__ bf16 to_bf16(float f) {
    union { float f; unsigned int u; } in; in.f = f;
    unsigned int u = in.u;
    u += 0x7fffu + ((u >> 16) & 1u);          // round-to-nearest-even
    union { unsigned short s; bf16 h; } out; out.s = (unsigned short)(u >> 16);
    return out.h;
}

#if USE_TDM
// 2D tile DMA: global (rows x 32 bf16, row stride = stride_elems) -> LDS
// (contiguous 64B rows). D# per CDNA5 ISA 8.3-8.6. OOB rows (>= rows_valid)
// read as zero.
__device__ __forceinline__ void tdm_load_2d(unsigned lds_off, const bf16* gptr,
                                            unsigned rows_valid, unsigned rows_tile,
                                            unsigned stride_elems)
{
    unsigned long long ga = (unsigned long long)(uintptr_t)gptr;
    const unsigned elems = 32;                     // tile_dim0 / tensor_dim0
    u32x4 g0;
    g0[0] = 1u;                                    // count=1, user descriptor
    g0[1] = lds_off;                               // lds_addr
    g0[2] = (unsigned)(ga & 0xFFFFFFFFu);          // global_addr[31:0]
    g0[3] = (unsigned)((ga >> 32) & 0x01FFFFFFu) | (2u << 30);  // [56:32] | type=2
    i32x8 g1;
    g1[0] = (int)(1u << 16);                       // data_size=1 (2B), no multicast
    g1[1] = (int)((elems & 0xFFFFu) << 16);        // tensor_dim0[15:0]
    g1[2] = (int)((elems >> 16) | ((rows_valid & 0xFFFFu) << 16)); // td0 hi | td1 lo
    g1[3] = (int)((rows_valid >> 16) | (elems << 16));            // td1 hi | tile_dim0
    g1[4] = (int)(rows_tile & 0xFFFFu);            // tile_dim1 (tile_dim2=0)
    g1[5] = (int)stride_elems;                     // tensor_dim0_stride[31:0]
    g1[6] = 0;                                     // stride0 hi | stride1 lo
    g1[7] = 0;
    i32x4 z4 = {0, 0, 0, 0};
    i32x8 z8 = {0, 0, 0, 0, 0, 0, 0, 0};
    // 6-arg form (clang-23 / therock-10.0): groups 2,3 + extra block + cpol
    __builtin_amdgcn_tensor_load_to_lds(g0, g1, z4, z4, z8, 0);
}
#endif

// ---------------------------------------------------------------------------
// d = rsqrt(colsum(A)) : 2D partial reduction, coalesced, then rsqrt in place
// ---------------------------------------------------------------------------
__global__ void colsum_partial(const float* __restrict__ A, float* __restrict__ acc,
                               int M, int rowsPer) {
    int c  = blockIdx.x * blockDim.x + threadIdx.x;
    int r0 = blockIdx.y * rowsPer;
    float s = 0.0f;
    for (int r = r0; r < r0 + rowsPer; ++r) s += A[(size_t)r * M + c];
    atomicAdd(&acc[c], s);
}

__global__ void rsqrt_inplace(float* __restrict__ d, int n) {
    int i = blockIdx.x * blockDim.x + threadIdx.x;
    if (i < n) d[i] = rsqrtf(d[i]);
}

// ---------------------------------------------------------------------------
// At_bf16[r][c] = bf16(d[r] * A[r][c] * d[c])   (8 elems / thread, vectorized)
// ---------------------------------------------------------------------------
__global__ void scale_cast_A(const float* __restrict__ A, const float* __restrict__ d,
                             bf16* __restrict__ At, int M) {
    size_t idx = ((size_t)blockIdx.x * blockDim.x + threadIdx.x) * 8;
    int r = (int)(idx / (size_t)M);
    int c = (int)(idx % (size_t)M);
    float dr = d[r];
    const float4* a4 = (const float4*)(A + idx);
    float4 a0 = a4[0], a1 = a4[1];
    const float4* d4 = (const float4*)(d + c);
    float4 d0 = d4[0], d1 = d4[1];
    bf16 o[8];
    o[0] = to_bf16(dr * a0.x * d0.x);  o[1] = to_bf16(dr * a0.y * d0.y);
    o[2] = to_bf16(dr * a0.z * d0.z);  o[3] = to_bf16(dr * a0.w * d0.w);
    o[4] = to_bf16(dr * a1.x * d1.x);  o[5] = to_bf16(dr * a1.y * d1.y);
    o[6] = to_bf16(dr * a1.z * d1.z);  o[7] = to_bf16(dr * a1.w * d1.w);
    *(uint4*)(At + idx) = *(const uint4*)o;
}

__global__ void cast_bf16(const float* __restrict__ src, bf16* __restrict__ dst, int n) {
    int i = (blockIdx.x * blockDim.x + threadIdx.x) * 4;
    if (i >= n) return;
    float4 v = *(const float4*)(src + i);
    dst[i + 0] = to_bf16(v.x); dst[i + 1] = to_bf16(v.y);
    dst[i + 2] = to_bf16(v.z); dst[i + 3] = to_bf16(v.w);
}

// dst[c][r] = bf16(src[r][c]);  src: [R x C] f32, dst: [C x R] bf16. 32x32 tiles.
__global__ void transpose_cast(const float* __restrict__ src, bf16* __restrict__ dst,
                               int R, int C) {
    __shared__ float t[32][33];
    int r0 = blockIdx.y * 32, c0 = blockIdx.x * 32;
    int tx = threadIdx.x, ty = threadIdx.y;          // block (32, 8)
    #pragma unroll
    for (int i = 0; i < 4; ++i)
        t[ty + i * 8][tx] = src[(size_t)(r0 + ty + i * 8) * C + c0 + tx];
    __syncthreads();
    #pragma unroll
    for (int i = 0; i < 4; ++i)
        dst[(size_t)(c0 + ty + i * 8) * R + r0 + tx] = to_bf16(t[tx][ty + i * 8]);
}

// ---------------------------------------------------------------------------
// WMMA GEMM: C = act(A @ B^T + bias)
//   A: [M x K] bf16 row-major,  B: [N x K] bf16 row-major
//   Cf (f32, row-major [M x N]) and Cb (bf16) written if non-null.
//   CBT=true  -> Cb stored transposed [N x M] (contiguous 16B stores per lane)
//   CBT=false -> Cb stored row-major  [M x N]
// Block tile 128x64, 8 waves, each wave a 32x32 tile (4 wmma frags).
// K-tiles staged by the Tensor Data Mover (double-buffered, TENSORcnt sync).
// M%128==0, K%32==0; N edge handled by TDM zero-fill / manual guard.
// ---------------------------------------------------------------------------
template<bool RELU, bool CBT>
__global__ void __launch_bounds__(256)
gemm_wmma(const bf16* __restrict__ A, const bf16* __restrict__ B,
          const float* __restrict__ bias,
          float* __restrict__ Cf, bf16* __restrict__ Cb,
          int M, int N, int K)
{
    __shared__ __align__(16) bf16 lA[2][128 * 32];   // [m][k], 8 KB per buffer
    __shared__ __align__(16) bf16 lB[2][64 * 32];    // [n][k], 4 KB per buffer

    const int tid  = threadIdx.x;
    const int lane = tid & 31;
    const int wave = tid >> 5;
    const int half = lane >> 4;
    const int lr   = lane & 15;

    const int m0 = blockIdx.x * 128;
    const int n0 = blockIdx.y * 64;
    const int wm = (wave >> 1) * 32;
    const int wn = (wave & 1) * 32;
    const unsigned nv = (N - n0 < 64) ? (unsigned)(N - n0) : 64u;

    const v8f vzero = {0.f, 0.f, 0.f, 0.f, 0.f, 0.f, 0.f, 0.f};
    v8f acc[2][2];
    acc[0][0] = vzero; acc[0][1] = vzero; acc[1][0] = vzero; acc[1][1] = vzero;

    const int steps = K >> 5;

#if USE_TDM
    auto issue = [&](int buf, int k0) {
        if (wave == 0) {
            tdm_load_2d((unsigned)(uintptr_t)&lA[buf][0],
                        A + (size_t)m0 * K + k0, 128u, 128u, (unsigned)K);
            tdm_load_2d((unsigned)(uintptr_t)&lB[buf][0],
                        B + (size_t)n0 * K + k0, nv, 64u, (unsigned)K);
        }
    };
    issue(0, 0);
#else
    auto stage = [&](int buf, int k0) {
        {   // A tile: 16 bf16 (2x uint4) per thread
            int row = tid >> 1;
            int kc  = (tid & 1) * 16;
            const uint4* src = (const uint4*)(A + (size_t)(m0 + row) * K + k0 + kc);
            uint4* dst = (uint4*)(&lA[buf][row * 32 + kc]);
            dst[0] = src[0];
            dst[1] = src[1];
        }
        {   // B tile: 8 bf16 (1x uint4) per thread, rows of B are contiguous
            int n  = tid >> 2;
            int kc = (tid & 3) * 8;
            uint4* dst = (uint4*)(&lB[buf][n * 32 + kc]);
            if ((unsigned)n < nv)
                *dst = *(const uint4*)(B + (size_t)(n0 + n) * K + k0 + kc);
            else
                *dst = make_uint4(0u, 0u, 0u, 0u);
        }
    };
    stage(0, 0);
#endif

    for (int i = 0; i < steps; ++i) {
        const int buf = i & 1;
#if USE_TDM
        if (wave == 0) {
            if (i + 1 < steps) {
                issue(buf ^ 1, (i + 1) * 32);
                __builtin_amdgcn_s_wait_tensorcnt(2);   // tile i complete
            } else {
                __builtin_amdgcn_s_wait_tensorcnt(0);   // drain
            }
        }
        __syncthreads();
#else
        __syncthreads();
#endif
        // ---- fragments (ISA 16-bit A-matrix 16x32 VGPR layout) ----
        v16bf afrag[2], bfrag[2];
        #pragma unroll
        for (int mi = 0; mi < 2; ++mi) {
            const bf16* pa = &lA[buf][(wm + mi * 16 + lr) * 32];
            #pragma unroll
            for (int j = 0; j < 8; ++j) {
                afrag[mi][j]     = pa[half * 8 + j];
                afrag[mi][8 + j] = pa[16 + half * 8 + j];
            }
        }
        #pragma unroll
        for (int ni = 0; ni < 2; ++ni) {
            const bf16* pb = &lB[buf][(wn + ni * 16 + lr) * 32];
            #pragma unroll
            for (int j = 0; j < 8; ++j) {
                bfrag[ni][j]     = pb[half * 8 + j];
                bfrag[ni][8 + j] = pb[16 + half * 8 + j];
            }
        }
        #pragma unroll
        for (int mi = 0; mi < 2; ++mi)
            #pragma unroll
            for (int ni = 0; ni < 2; ++ni)
                acc[mi][ni] = __builtin_amdgcn_wmma_f32_16x16x32_bf16(
                    false, afrag[mi], false, bfrag[ni],
                    (short)0, acc[mi][ni], false, false);
#if !USE_TDM
        __syncthreads();
        if (i + 1 < steps) stage(buf ^ 1, (i + 1) * 32);
#else
        __syncthreads();   // all reads of buf done before TDM overwrites it
#endif
    }

    // ---- epilogue: bias + ReLU; f32 row-major, bf16 row-major or transposed ----
    #pragma unroll
    for (int mi = 0; mi < 2; ++mi) {
        #pragma unroll
        for (int ni = 0; ni < 2; ++ni) {
            int n = n0 + wn + ni * 16 + lr;
            if (n < N) {
                float bv = bias ? bias[n] : 0.0f;
                int mb = m0 + wm + mi * 16 + half * 8;
                float cv[8];
                #pragma unroll
                for (int v = 0; v < 8; ++v) {
                    float c = acc[mi][ni][v] + bv;
                    if (RELU) c = fmaxf(c, 0.0f);
                    cv[v] = c;
                }
                if (Cf) {
                    #pragma unroll
                    for (int v = 0; v < 8; ++v)
                        Cf[(size_t)(mb + v) * N + n] = cv[v];
                }
                if (Cb) {
                    if (CBT) {        // transposed: 8 consecutive bf16 -> one 16B store
                        bf16 o[8];
                        #pragma unroll
                        for (int v = 0; v < 8; ++v) o[v] = to_bf16(cv[v]);
                        *(uint4*)(Cb + (size_t)n * M + mb) = *(const uint4*)o;
                    } else {
                        #pragma unroll
                        for (int v = 0; v < 8; ++v)
                            Cb[(size_t)(mb + v) * N + n] = to_bf16(cv[v]);
                    }
                }
            }
        }
    }
}

// ---------------------------------------------------------------------------
// Orchestration
// ---------------------------------------------------------------------------
extern "C" void kernel_launch(void* const* d_in, const int* in_sizes, int n_in,
                              void* d_out, int out_size, void* d_ws, size_t ws_size,
                              hipStream_t stream)
{
    const int M = 8192, NIN = 512, NOUT = 512, NCLS = 32;
    const float* x   = (const float*)d_in[0];
    const float* A   = (const float*)d_in[1];
    const float* W1w = (const float*)d_in[2];
    const float* W1b = (const float*)d_in[3];
    const float* W2w = (const float*)d_in[4];
    const float* W2b = (const float*)d_in[5];
    const float* FLw = (const float*)d_in[6];
    const float* FLb = (const float*)d_in[7];

    float* outH = (float*)d_out;                    // h2: (M, NOUT) f32
    float* outL = outH + (size_t)M * NOUT;          // logits: (M, NCLS) f32

    char* ws = (char*)d_ws;
    float* dvec = (float*)ws;  ws += (size_t)M * sizeof(float);
    bf16* Atb  = (bf16*)ws;    ws += (size_t)M * M * sizeof(bf16);      // 128 MB
    bf16* xT   = (bf16*)ws;    ws += (size_t)NIN * M * sizeof(bf16);    // x^T, reused as h2 bf16
    bf16* yrm  = (bf16*)ws;    ws += (size_t)M * NOUT * sizeof(bf16);   // y1/y2 row-major
    bf16* h1T  = (bf16*)ws;    ws += (size_t)NOUT * M * sizeof(bf16);   // h1 transposed
    bf16* W1bb = (bf16*)ws;    ws += (size_t)NOUT * NIN * sizeof(bf16);
    bf16* W2bb = (bf16*)ws;    ws += (size_t)NOUT * NOUT * sizeof(bf16);
    bf16* FLbb = (bf16*)ws;    ws += (size_t)NCLS * NOUT * sizeof(bf16);
    bf16* h2rm = xT;           // xT is dead once y1 is computed

    dim3 b256(256);

    // d = rsqrt(colsum(A))
    (void)hipMemsetAsync(dvec, 0, (size_t)M * sizeof(float), stream);
    colsum_partial<<<dim3(M / 256, 32), b256, 0, stream>>>(A, dvec, M, M / 32);
    rsqrt_inplace<<<M / 256, b256, 0, stream>>>(dvec, M);

    // At = bf16(d[:,None] * A * d[None,:]);  xT = bf16(x)^T; weight casts
    scale_cast_A<<<(unsigned)((size_t)M * M / (256 * 8)), b256, 0, stream>>>(A, dvec, Atb, M);
    transpose_cast<<<dim3(NIN / 32, M / 32), dim3(32, 8), 0, stream>>>(x, xT, M, NIN);
    cast_bf16<<<(NOUT * NIN / 4 + 255) / 256, b256, 0, stream>>>(W1w, W1bb, NOUT * NIN);
    cast_bf16<<<(NOUT * NOUT / 4 + 255) / 256, b256, 0, stream>>>(W2w, W2bb, NOUT * NOUT);
    cast_bf16<<<(NCLS * NOUT / 4 + 255) / 256, b256, 0, stream>>>(FLw, FLbb, NCLS * NOUT);

    // y1 = At @ x = At @ (xT)^T          (8192x8192x512 bf16 WMMA, f32 accum)
    gemm_wmma<false, false><<<dim3(M / 128, NIN / 64), b256, 0, stream>>>(
        Atb, xT, nullptr, nullptr, yrm, M, NIN, M);
    // h1 = relu(y1 @ W1^T + b1) -> h1T (transposed bf16, contiguous stores)
    gemm_wmma<true, true><<<dim3(M / 128, NOUT / 64), b256, 0, stream>>>(
        yrm, W1bb, W1b, nullptr, h1T, M, NOUT, NIN);
    // y2 = At @ h1 = At @ (h1T)^T        (At ~L2-resident on this second pass)
    gemm_wmma<false, false><<<dim3(M / 128, NOUT / 64), b256, 0, stream>>>(
        Atb, h1T, nullptr, nullptr, yrm, M, NOUT, M);
    // h2 = relu(y2 @ W2^T + b2) -> f32 d_out + bf16 row-major feed
    gemm_wmma<true, false><<<dim3(M / 128, NOUT / 64), b256, 0, stream>>>(
        yrm, W2bb, W2b, outH, h2rm, M, NOUT, NOUT);
    // logits = h2 @ FL^T + bFL           (N=32 edge: TDM zero-fill / guard)
    gemm_wmma<false, false><<<dim3(M / 128, 1), b256, 0, stream>>>(
        h2rm, FLbb, FLb, outL, nullptr, M, NCLS, NOUT);
}
